// SelfAttention_36215164240477
// MI455X (gfx1250) — compile-verified, matrix operand order checked by reading, not run on hardware
//
#include <hip/hip_runtime.h>

// ---------------------------------------------------------------------------
// Self-attention forward on MI455X (gfx1250).
//   All GEMMs: v_wmma_f32_16x16x32_f16 (fp32 accumulate).
//   QKV GEMM: LDS double-buffered, async global->LDS staging (ASYNCcnt waits).
//   Attention: Q block staged once into 258KB of WGP LDS (padded, conflict-free
//   fragment reads); K/V streamed from L2 with prefetch hints; two-pass
//   (stats then output) so the 256MB score matrix never touches HBM.
// ---------------------------------------------------------------------------

typedef __attribute__((ext_vector_type(16))) _Float16 v16h;
typedef __attribute__((ext_vector_type(8)))  float    v8f;
typedef int v4i __attribute__((vector_size(16)));   // matches async-LDS builtin

#define WMMA_F16(a, b, c) \
  __builtin_amdgcn_wmma_f32_16x16x32_f16(false, (a), false, (b), (short)0, (c), false, false)

#if defined(__has_builtin)
#  if __has_builtin(__builtin_amdgcn_global_load_async_to_lds_b128)
#    define USE_ASYNC 1
#  endif
#endif
#ifndef USE_ASYNC
#  define USE_ASYNC 0
#endif

#if USE_ASYNC
#  if __has_builtin(__builtin_amdgcn_s_wait_asynccnt)
#    define WAIT_ASYNC(n) __builtin_amdgcn_s_wait_asynccnt(n)
#  else
#    define WAIT_ASYNC(n) asm volatile("s_wait_asynccnt %0" ::"i"(n) : "memory")
#  endif
#else
#  define WAIT_ASYNC(n)
#endif

// Copy 16B global -> LDS (async DMA if available, else load+ds_store).
__device__ __forceinline__ void cp16(const _Float16* g, _Float16* l) {
#if USE_ASYNC
  __builtin_amdgcn_global_load_async_to_lds_b128(
      (__attribute__((address_space(1))) v4i*)(g),
      (__attribute__((address_space(3))) v4i*)(l), 0, 0);
#else
  *reinterpret_cast<uint4*>(l) = *reinterpret_cast<const uint4*>(g);
#endif
}

union FragU { v16h v; uint4 q[2]; };

// A/B fragment from a row-major [rows][ldm] f16 matrix (global).
// Lane l<16: row0+l, K = kk+[0..7],[16..23]; lane>=16: same rows, K +8.
__device__ __forceinline__ v16h load_frag(const _Float16* __restrict__ base,
                                          int row0, int ldm, int kk, int lane) {
  int r  = row0 + (lane & 15);
  int kb = kk + ((lane >> 4) << 3);
  const _Float16* p = base + (size_t)r * ldm + kb;
  FragU u;
  u.q[0] = *reinterpret_cast<const uint4*>(p);
  u.q[1] = *reinterpret_cast<const uint4*>(p + 16);
  return u.v;
}

// Same fragment pattern from an LDS tile with row stride `ldl` (f16 units).
__device__ __forceinline__ v16h frag_lds(const _Float16* tile, int row0, int ldl,
                                         int kk, int lane) {
  const _Float16* p = tile + (row0 + (lane & 15)) * ldl + kk + ((lane >> 4) << 3);
  FragU u;
  u.q[0] = *reinterpret_cast<const uint4*>(p);
  u.q[1] = *reinterpret_cast<const uint4*>(p + 16);
  return u.v;
}

// ------------------------------ Pass 0 -------------------------------------
__global__ __launch_bounds__(256) void cvt_x_f16(const float* __restrict__ X,
                                                 _Float16* __restrict__ Xh) {
  size_t i = (size_t)blockIdx.x * 256 + threadIdx.x;
  Xh[i] = (_Float16)X[i];
}

__global__ __launch_bounds__(256) void cvt_w_f16(const float* __restrict__ Wq,
                                                 const float* __restrict__ Wk,
                                                 const float* __restrict__ Wv,
                                                 _Float16* __restrict__ Wt) {
  int i = blockIdx.x * 256 + threadIdx.x;   // 0 .. 1024*1024-1
  int w = blockIdx.y;
  const float* W = (w == 0) ? Wq : (w == 1) ? Wk : Wv;
  int e = i >> 10, d = i & 1023;
  Wt[(size_t)w * 1048576 + i] = (_Float16)W[(size_t)d * 1024 + e];
}

// ------------------------------ Pass 1: QKV --------------------------------
// 128x128 block tile, LDS double-buffered async staging, wave tile 32x64.
__global__ __launch_bounds__(256) void qkv_gemm(
    const _Float16* __restrict__ Xh,   // [16384][1024]
    const _Float16* __restrict__ Wt,   // [3][1024 e][1024 d]
    const float* __restrict__ bq, const float* __restrict__ bk,
    const float* __restrict__ bv,
    _Float16* __restrict__ Qh, _Float16* __restrict__ Kh,
    _Float16* __restrict__ Vt)         // [4][1024 e][4096 k]
{
  __shared__ _Float16 Xs[2][128 * 40];   // 32-col tiles padded to 40 (80B rows)
  __shared__ _Float16 Ws[2][128 * 40];
  const int wsel = blockIdx.z;
  const int tid = threadIdx.x, wave = tid >> 5, lane = tid & 31;
  const int mblk = blockIdx.y * 128, nblk = blockIdx.x * 128;
  const int m_off = (wave >> 1) * 32, n_off = (wave & 1) * 64;
  const _Float16* W = Wt + (size_t)wsel * 1048576;
  const float* bias = (wsel == 0) ? bq : (wsel == 1) ? bk : bv;

  auto stage = [&](int buf, int kk) {
#pragma unroll
    for (int i = 0; i < 2; ++i) {
      int c = tid + 256 * i;           // 512 16B chunks per 128x32 tile
      int row = c >> 2, cb = (c & 3) * 8;
      cp16(Xh + (size_t)(mblk + row) * 1024 + kk + cb, &Xs[buf][row * 40 + cb]);
      cp16(W  + (size_t)(nblk + row) * 1024 + kk + cb, &Ws[buf][row * 40 + cb]);
    }
  };

  stage(0, 0);
  stage(1, 32);

  v8f acc[2][4] = {};
  for (int it = 0; it < 32; ++it) {
    const int buf = it & 1;
#if USE_ASYNC
    if (it == 31) { WAIT_ASYNC(0); } else { WAIT_ASYNC(4); }  // retire stage(it)
#endif
    __syncthreads();
    v16h a[2], b[4];
#pragma unroll
    for (int mi = 0; mi < 2; ++mi) a[mi] = frag_lds(Xs[buf], m_off + 16 * mi, 40, 0, lane);
#pragma unroll
    for (int ni = 0; ni < 4; ++ni) b[ni] = frag_lds(Ws[buf], n_off + 16 * ni, 40, 0, lane);
#pragma unroll
    for (int mi = 0; mi < 2; ++mi)
#pragma unroll
      for (int ni = 0; ni < 4; ++ni)
        acc[mi][ni] = WMMA_F16(a[mi], b[ni], acc[mi][ni]);
    __syncthreads();                         // everyone done reading `buf`
    if (it + 2 < 32) stage(buf, (it + 2) * 32);
  }

  const int lh = lane >> 4, lc = lane & 15;
#pragma unroll
  for (int mi = 0; mi < 2; ++mi)
#pragma unroll
    for (int ni = 0; ni < 4; ++ni) {
      int col = nblk + n_off + 16 * ni + lc;
      float bb = bias[col];
#pragma unroll
      for (int r = 0; r < 8; ++r) {
        int row = mblk + m_off + 16 * mi + r + 8 * lh;
        _Float16 h = (_Float16)(acc[mi][ni][r] + bb);
        if (wsel == 0)      Qh[(size_t)row * 1024 + col] = h;
        else if (wsel == 1) Kh[(size_t)row * 1024 + col] = h;
        else {
          int batch = row >> 12, k = row & 4095;
          Vt[((size_t)batch * 1024 + col) * 4096 + k] = h;
        }
      }
    }
}

// ------------------------------ Pass 2: stats ------------------------------
// 128 q-rows per block staged once into LDS (wave-private 16-row slices).
__global__ __launch_bounds__(256) void attn_stats(
    const _Float16* __restrict__ Qh, const _Float16* __restrict__ Kh,
    float* __restrict__ rowmax, float* __restrict__ rowsum)
{
  __shared__ _Float16 Qs[128 * 1032];        // 258KB of the 320KB WGP LDS
  const int tid = threadIdx.x, wave = tid >> 5, lane = tid & 31;
  const int qrow0 = blockIdx.x * 128;
  const int batch = qrow0 >> 12;
  const _Float16* Kb = Kh + ((size_t)batch << 12) * 1024;
  const float scale = 0.03125f;

  {  // wave-private staging: rows [16*wave, 16*wave+16)
    const _Float16* src = Qh + (size_t)(qrow0 + wave * 16) * 1024;
    _Float16* dst = Qs + wave * 16 * 1032;
    for (int i = 0; i < 64; ++i) {
      int c = lane + 32 * i;                 // 2048 chunks of 16B
      int row = c >> 7, cb = (c & 127) * 8;
      cp16(src + (size_t)row * 1024 + cb, dst + row * 1032 + cb);
    }
    WAIT_ASYNC(0);
  }  // no barrier needed: LDS region is wave-private, LDS ops in-order per wave

  const _Float16* qa = Qs + wave * 16 * 1032;
  float m[8], l[8];
#pragma unroll
  for (int r = 0; r < 8; ++r) { m[r] = -1e30f; l[r] = 0.f; }

  for (int kt = 0; kt < 4096; kt += 16) {
    if (kt + 16 < 4096) __builtin_prefetch(Kb + (size_t)(kt + 16) * 1024, 0, 1);
    v8f s = {};
    for (int e = 0; e < 1024; e += 32) {
      v16h a = frag_lds(qa, 0, 1032, e, lane);
      v16h b = load_frag(Kb, kt, 1024, e, lane);
      s = WMMA_F16(a, b, s);
    }
#pragma unroll
    for (int r = 0; r < 8; ++r) {
      float v = s[r] * scale;
      float tm = v;
#pragma unroll
      for (int off = 1; off < 16; off <<= 1) tm = fmaxf(tm, __shfl_xor(tm, off, 32));
      float mn = fmaxf(m[r], tm);
      float ex = __expf(v - mn);
#pragma unroll
      for (int off = 1; off < 16; off <<= 1) ex += __shfl_xor(ex, off, 32);
      l[r] = l[r] * __expf(m[r] - mn) + ex;
      m[r] = mn;
    }
  }
  if ((lane & 15) == 0) {
#pragma unroll
    for (int r = 0; r < 8; ++r) {
      int row = qrow0 + wave * 16 + r + 8 * (lane >> 4);
      rowmax[row] = m[r];
      rowsum[row] = l[r];
    }
  }
}

// ------------------------------ Pass 3: output -----------------------------
// Block: 128q x 128e. Q staged once in LDS; per 32-key step recompute S (WMMA),
// P=exp(S-m) via wave-private LDS repack, P·V via WMMA against Vt.
__global__ __launch_bounds__(256) void attn_out(
    const _Float16* __restrict__ Qh, const _Float16* __restrict__ Kh,
    const _Float16* __restrict__ Vt,
    const float* __restrict__ rowmax, const float* __restrict__ rowsum,
    float* __restrict__ Out)
{
  __shared__ _Float16 Qs[128 * 1032];        // 258KB
  __shared__ _Float16 Psh[8][1024];          // 16KB, wave-private P tiles
  const int tid = threadIdx.x, wave = tid >> 5, lane = tid & 31;
  const int lh = lane >> 4, lc = lane & 15;
  const int batch = blockIdx.z;
  const int q0blk = blockIdx.y * 128;
  const int m_off = (wave >> 1) * 32;
  const int e0 = blockIdx.x * 128 + (wave & 1) * 64;
  const _Float16* Qb = Qh + ((size_t)batch << 12) * 1024 + (size_t)q0blk * 1024;
  const _Float16* Kb = Kh + ((size_t)batch << 12) * 1024;
  const _Float16* Vb = Vt + (size_t)batch * 1024 * 4096;
  const float* rm = rowmax + (batch << 12);
  const float* rs = rowsum + (batch << 12);
  const float scale = 0.03125f;

  // cooperative staging of all 128 Q rows
  for (int i = 0; i < 64; ++i) {
    int c = tid + 256 * i;                   // 16384 chunks of 16B
    int row = c >> 7, cb = (c & 127) * 8;
    cp16(Qb + (size_t)row * 1024 + cb, Qs + row * 1032 + cb);
  }
  WAIT_ASYNC(0);
  __syncthreads();

  float mrow[2][8], rinv[2][8];
#pragma unroll
  for (int mi = 0; mi < 2; ++mi)
#pragma unroll
    for (int r = 0; r < 8; ++r) {
      int row = q0blk + m_off + 16 * mi + r + 8 * lh;
      mrow[mi][r] = rm[row];
      rinv[mi][r] = 1.0f / rs[row];
    }

  v8f o[2][4] = {};
  _Float16* P = Psh[wave];

  for (int kt = 0; kt < 4096; kt += 32) {
    if (kt + 32 < 4096) {
      __builtin_prefetch(Kb + (size_t)(kt + 32) * 1024, 0, 1);
      __builtin_prefetch(Vb + (size_t)e0 * 4096 + kt + 32, 0, 1);
    }
    // ---- S tile (32q x 32k), full e reduction; Q frags from LDS ----
    v8f s[2][2] = {};
    for (int e = 0; e < 1024; e += 32) {
      v16h a0 = frag_lds(Qs, m_off, 1032, e, lane);
      v16h a1 = frag_lds(Qs, m_off + 16, 1032, e, lane);
      v16h b0 = load_frag(Kb, kt,      1024, e, lane);
      v16h b1 = load_frag(Kb, kt + 16, 1024, e, lane);
      s[0][0] = WMMA_F16(a0, b0, s[0][0]);
      s[0][1] = WMMA_F16(a0, b1, s[0][1]);
      s[1][0] = WMMA_F16(a1, b0, s[1][0]);
      s[1][1] = WMMA_F16(a1, b1, s[1][1]);
    }
    // ---- P = exp(scale*S - m) into wave-private LDS ----
#pragma unroll
    for (int mi = 0; mi < 2; ++mi)
#pragma unroll
      for (int ki = 0; ki < 2; ++ki)
#pragma unroll
        for (int r = 0; r < 8; ++r) {
          float p = __expf(s[mi][ki][r] * scale - mrow[mi][r]);
          P[(16 * mi + r + 8 * lh) * 32 + 16 * ki + lc] = (_Float16)p;
        }
    // ---- P·V ----
    v16h pa[2];
#pragma unroll
    for (int mi = 0; mi < 2; ++mi) pa[mi] = frag_lds(P, 16 * mi, 32, 0, lane);
    v16h vbf[4];
#pragma unroll
    for (int ni = 0; ni < 4; ++ni) vbf[ni] = load_frag(Vb, e0 + 16 * ni, 4096, kt, lane);
#pragma unroll
    for (int mi = 0; mi < 2; ++mi)
#pragma unroll
      for (int ni = 0; ni < 4; ++ni)
        o[mi][ni] = WMMA_F16(pa[mi], vbf[ni], o[mi][ni]);
  }

#pragma unroll
  for (int mi = 0; mi < 2; ++mi)
#pragma unroll
    for (int ni = 0; ni < 4; ++ni)
#pragma unroll
      for (int r = 0; r < 8; ++r) {
        int row = q0blk + m_off + 16 * mi + r + 8 * lh;
        int col = e0 + 16 * ni + lc;
        Out[(((size_t)batch << 12) + row) * 1024 + col] = o[mi][ni][r] * rinv[mi][r];
      }
}

// ------------------------------ launcher -----------------------------------
extern "C" void kernel_launch(void* const* d_in, const int* in_sizes, int n_in,
                              void* d_out, int out_size, void* d_ws, size_t ws_size,
                              hipStream_t stream) {
  const float* X  = (const float*)d_in[0];
  const float* Wq = (const float*)d_in[1];
  const float* Wk = (const float*)d_in[2];
  const float* Wv = (const float*)d_in[3];
  const float* bq = (const float*)d_in[4];
  const float* bk = (const float*)d_in[5];
  const float* bv = (const float*)d_in[6];
  float* Out = (float*)d_out;

  char* ws = (char*)d_ws;
  _Float16* Xh = (_Float16*)ws;  ws += (size_t)16384 * 1024 * 2;  // 32 MB
  _Float16* Wt = (_Float16*)ws;  ws += (size_t)3 * 1024 * 1024 * 2;
  _Float16* Qh = (_Float16*)ws;  ws += (size_t)16384 * 1024 * 2;
  _Float16* Kh = (_Float16*)ws;  ws += (size_t)16384 * 1024 * 2;
  _Float16* Vt = (_Float16*)ws;  ws += (size_t)16384 * 1024 * 2;
  float* rowmax = (float*)ws;    ws += (size_t)16384 * 4;
  float* rowsum = (float*)ws;

  cvt_x_f16<<<16384 * 1024 / 256, 256, 0, stream>>>(X, Xh);
  cvt_w_f16<<<dim3(4096, 3), 256, 0, stream>>>(Wq, Wk, Wv, Wt);
  qkv_gemm<<<dim3(8, 128, 3), 256, 0, stream>>>(Xh, Wt, bq, bk, bv, Qh, Kh, Vt);
  attn_stats<<<128, 256, 0, stream>>>(Qh, Kh, rowmax, rowsum);
  attn_out<<<dim3(8, 32, 4), 256, 0, stream>>>(Qh, Kh, Vt, rowmax, rowsum, Out);
}
